// ObjectAttentionBlock_1176821039805
// MI455X (gfx1250) — matrix operand order repaired and checked by reading, not compiled, not run on hardware
//
#include <hip/hip_runtime.h>
#include <hip/hip_bf16.h>

// ---------------------------------------------------------------------------
// Fused ObjectAttentionBlock for gfx1250 (MI455X, wave32, WMMA).
//   q  = relu(BN(Wp1 x)); q = relu(BN(Wp2 q))
//   k  = relu(BN(Wo2 relu(BN(Wo1 proxy)))); v = relu(BN(Wd proxy))
//   ctx = softmax(q^T k / sqrt(512)) v ; out = relu(BN(Wu ctx))
// Heavy 512x512 GEMMs run on v_wmma_f32_16x16x32_f16 with BN folded into f16
// weights; entire pixel chain fused per 32-pixel tile in LDS.
// ---------------------------------------------------------------------------

typedef __attribute__((ext_vector_type(16))) _Float16 v16h;
typedef __attribute__((ext_vector_type(2)))  _Float16 h2;
typedef __attribute__((ext_vector_type(8)))  float    v8f;

#define CC      512
#define HW_     16384
#define NBATCH  8
#define KOBJ    19
#define TP      32                      // pixels per block tile
#define CPAD    514                     // padded channel stride (bank spread)
#define BUFB    (TP * CPAD * 2)         // 32,896 B per f16 tile buffer
#define SIMPAD  20
#define SMEM_BYTES (3 * BUFB + TP * SIMPAD * 4)   // 101,248 B
#define OPAD    33                      // output staging [512][33] f32

// ---------------- WMMA fragment loaders (contiguous-k layouts, 16x16x32 f16) --------

// A: 16x32 (MxK). lanes 0-15: M=lane, VGPR0..7 hold K=0..15.
//    lanes 16-31: same M, K=16..31 (matches contiguous B layout).
__device__ __forceinline__ v16h load_a_frag(const _Float16* xs, int mbase,
                                            int k0, int lane) {
  const int m    = mbase + (lane & 15);
  const int koff = (lane < 16) ? 0 : 16;
  const _Float16* row = xs + m * CPAD;
  v16h a;
#pragma unroll
  for (int i = 0; i < 8; ++i) {
    const int kk = k0 + koff + 2 * i;
    h2 p = *(const h2*)(row + kk);
    a[2 * i]     = p.x;
    a[2 * i + 1] = p.y;
  }
  return a;
}

// B: 32x16 (KxN), N = output channel. W is folded weight, row-major [N][K]:
// lane n = lane&15, lanes 0-15 read K=k0..k0+15, lanes 16-31 K=k0+16..k0+31
// -> one contiguous 32-byte read per lane.
__device__ __forceinline__ v16h load_b_frag(const _Float16* W, int nbase,
                                            int k0, int lane) {
  const int n  = nbase + (lane & 15);
  const int kk = k0 + ((lane < 16) ? 0 : 16);
  return *(const v16h*)(W + (size_t)n * CC + kk);
}

// ---------------- wave-level GEMM: Mtile=16, 8 Ntiles, K=512 ----------------
// 8 waves/block cover M=32 (w&1) x N=512 ((w>>1)*8 ntiles).
template <bool TO_HALF>
__device__ __forceinline__ void wave_gemm512(const _Float16* Asrc,
                                             const _Float16* W,
                                             const float* bias,
                                             _Float16* dst_h, float* dst_f,
                                             int tid) {
  const int lane   = tid & 31;
  const int w      = tid >> 5;
  const int mbase  = (w & 1) * 16;
  const int ntbase = (w >> 1) * 8;

  v8f acc[8] = {};

  for (int s = 0; s < 16; ++s) {
    const int k0 = s * 32;
    const v16h a = load_a_frag(Asrc, mbase, k0, lane);
    // prefetch next K-slab of this wave's weight stream (L2 temporal)
    if (s < 15)
      __builtin_prefetch(W + (size_t)(ntbase * 16 + (lane & 15)) * CC + k0 + 32, 0, 1);
#pragma unroll
    for (int i = 0; i < 8; ++i) {
      const v16h b = load_b_frag(W, (ntbase + i) * 16, k0, lane);
      acc[i] = __builtin_amdgcn_wmma_f32_16x16x32_f16(
          false, a, false, b, (short)0, acc[i], false, false);
    }
  }

  const int nlane = lane & 15;
  const int mhi   = (lane < 16) ? 0 : 8;

  if constexpr (!TO_HALF) {
    // output staging aliases earlier (now dead) buffers: sync before stores
    __syncthreads();
  }

#pragma unroll
  for (int i = 0; i < 8; ++i) {
    const int nch = (ntbase + i) * 16 + nlane;
    const float bb = bias[nch];
#pragma unroll
    for (int r = 0; r < 8; ++r) {
      float val = acc[i][r] + bb;
      val = val > 0.f ? val : 0.f;
      const int m = mbase + mhi + r;
      if constexpr (TO_HALF)
        dst_h[m * CPAD + nch] = (_Float16)val;
      else
        dst_f[nch * OPAD + m] = val;
    }
  }
}

// ---------------- main fused kernel ----------------------------------------
__global__ __launch_bounds__(256) void oab_main(
    const float* __restrict__ x,
    const _Float16* __restrict__ wh1, const float* __restrict__ b1,
    const _Float16* __restrict__ wh2, const float* __restrict__ b2,
    const _Float16* __restrict__ whu, const float* __restrict__ bu,
    const float* __restrict__ kbuf,   // [n][19][512]
    const float* __restrict__ vbuf,   // [n][512][19]
    float* __restrict__ out) {
  extern __shared__ char smem[];
  _Float16* xs  = (_Float16*)(smem);             // buf0: x tile  [32][CPAD]
  _Float16* q1  = (_Float16*)(smem + BUFB);      // buf1: q1 / later free
  _Float16* q2  = (_Float16*)(smem + 2 * BUFB);  // buf2: q2, then ctx
  float*    simb = (float*)(smem + 3 * BUFB);    // [32][20]
  float*    outb = (float*)(smem);               // [512][33] (after barrier)

  const int tid = threadIdx.x;
  const size_t gp0  = (size_t)blockIdx.x * TP;
  const int    nb   = (int)(gp0 / HW_);
  const int    p_in = (int)(gp0 % HW_);
  const float* xbase = x + (size_t)nb * CC * HW_ + p_in;

  // stage x tile -> LDS f16, [pixel][channel]; coalesced along w
  for (int idx = tid; idx < TP * CC; idx += 256) {
    const int c = idx >> 5;
    const int p = idx & 31;
    xs[p * CPAD + c] = (_Float16)xbase[(size_t)c * HW_ + p];
  }
  __syncthreads();

  // f_pixel conv1 + BN + ReLU
  wave_gemm512<true>(xs, wh1, b1, q1, nullptr, tid);
  __syncthreads();
  // f_pixel conv2 + BN + ReLU
  wave_gemm512<true>(q1, wh2, b2, q2, nullptr, tid);
  __syncthreads();

  // sim[p][j] = (q2[p,:] . k[n,j,:]) / sqrt(512)
  const float* kb = kbuf + (size_t)nb * KOBJ * CC;
  for (int idx = tid; idx < TP * KOBJ; idx += 256) {
    const int p = idx / KOBJ;
    const int j = idx - p * KOBJ;
    const float*     kr = kb + j * CC;
    const _Float16*  qr = q2 + p * CPAD;
    float s = 0.f;
#pragma unroll 8
    for (int c = 0; c < CC; ++c) s += (float)qr[c] * kr[c];
    simb[p * SIMPAD + j] = s * 0.04419417382415922f;  // 512^-0.5
  }
  __syncthreads();

  // softmax over 19 objects, one thread per pixel
  if (tid < TP) {
    float mx = -1e30f;
#pragma unroll
    for (int j = 0; j < KOBJ; ++j) mx = fmaxf(mx, simb[tid * SIMPAD + j]);
    float e[KOBJ];
    float sum = 0.f;
#pragma unroll
    for (int j = 0; j < KOBJ; ++j) {
      e[j] = __expf(simb[tid * SIMPAD + j] - mx);
      sum += e[j];
    }
    const float inv = 1.f / sum;
#pragma unroll
    for (int j = 0; j < KOBJ; ++j) simb[tid * SIMPAD + j] = e[j] * inv;
  }
  __syncthreads();

  // ctx[p][c] = sum_j attn[p][j] * v[n][c][j]  (overwrite q2 buffer, f16)
  const float* vb = vbuf + (size_t)nb * CC * KOBJ;
  for (int idx = tid; idx < TP * CC; idx += 256) {
    const int c = idx >> 5;
    const int p = idx & 31;
    const float* vr = vb + c * KOBJ;
    const float* ar = simb + p * SIMPAD;
    float s = 0.f;
#pragma unroll
    for (int j = 0; j < KOBJ; ++j) s += ar[j] * vr[j];
    q2[p * CPAD + c] = (_Float16)s;
  }
  __syncthreads();

  // f_up conv + BN + ReLU -> staged to [channel][pixel] f32 (aliases smem 0)
  wave_gemm512<false>(q2, whu, bu, nullptr, outb, tid);
  __syncthreads();

  // coalesced NCHW writeback
  float* obase = out + (size_t)nb * CC * HW_ + p_in;
  for (int idx = tid; idx < TP * CC; idx += 256) {
    const int c = idx >> 5;
    const int p = idx & 31;
    obase[(size_t)c * HW_ + p] = outb[c * OPAD + p];
  }
}

// ---------------- prologue: fold BN into f16 weights ------------------------
__global__ void fold_weights(const float* __restrict__ w, const float* __restrict__ g,
                             const float* __restrict__ b, const float* __restrict__ m,
                             const float* __restrict__ v, _Float16* __restrict__ wh,
                             float* __restrict__ bias) {
  const int idx = blockIdx.x * 256 + threadIdx.x;
  if (idx >= CC * CC) return;
  const int o = idx / CC;
  const float s = g[o] * rsqrtf(v[o] + 1e-5f);
  wh[idx] = (_Float16)(w[idx] * s);
  if ((idx % CC) == 0) bias[o] = b[o] - m[o] * s;
}

// ---------------- prologue: proxy (object) path, f32 scalar -----------------
// h1[n][j][o] = relu(BN(Wo1 proxy))   proxy layout: [(n*512+c)*19 + j]
__global__ void proxy_stage1(const float* __restrict__ proxy, const float* __restrict__ w,
                             const float* __restrict__ g, const float* __restrict__ b,
                             const float* __restrict__ m, const float* __restrict__ v,
                             float* __restrict__ h1) {
  const int idx = blockIdx.x * 256 + threadIdx.x;
  if (idx >= NBATCH * KOBJ * CC) return;
  const int o  = idx & (CC - 1);
  const int j  = (idx >> 9) % KOBJ;
  const int nn = idx / (CC * KOBJ);
  const float* wr = w + (size_t)o * CC;
  const float* pr = proxy + (size_t)nn * CC * KOBJ + j;
  float s = 0.f;
  for (int c = 0; c < CC; ++c) s += wr[c] * pr[(size_t)c * KOBJ];
  const float sc  = g[o] * rsqrtf(v[o] + 1e-5f);
  const float val = s * sc + (b[o] - m[o] * sc);
  h1[((size_t)nn * KOBJ + j) * CC + o] = fmaxf(val, 0.f);
}

// kb[n][j][o] = relu(BN(Wo2 h1))   h1 contiguous in c
__global__ void proxy_stage2(const float* __restrict__ h1, const float* __restrict__ w,
                             const float* __restrict__ g, const float* __restrict__ b,
                             const float* __restrict__ m, const float* __restrict__ v,
                             float* __restrict__ kb) {
  const int idx = blockIdx.x * 256 + threadIdx.x;
  if (idx >= NBATCH * KOBJ * CC) return;
  const int o  = idx & (CC - 1);
  const int j  = (idx >> 9) % KOBJ;
  const int nn = idx / (CC * KOBJ);
  const float* wr = w + (size_t)o * CC;
  const float* hr = h1 + ((size_t)nn * KOBJ + j) * CC;
  float s = 0.f;
  for (int c = 0; c < CC; ++c) s += wr[c] * hr[c];
  const float sc  = g[o] * rsqrtf(v[o] + 1e-5f);
  const float val = s * sc + (b[o] - m[o] * sc);
  kb[((size_t)nn * KOBJ + j) * CC + o] = fmaxf(val, 0.f);
}

// vb[n][o][j] = relu(BN(Wd proxy))
__global__ void proxy_value(const float* __restrict__ proxy, const float* __restrict__ w,
                            const float* __restrict__ g, const float* __restrict__ b,
                            const float* __restrict__ m, const float* __restrict__ v,
                            float* __restrict__ vb) {
  const int idx = blockIdx.x * 256 + threadIdx.x;
  if (idx >= NBATCH * CC * KOBJ) return;
  const int j  = idx % KOBJ;
  const int o  = (idx / KOBJ) & (CC - 1);
  const int nn = idx / (KOBJ * CC);
  const float* wr = w + (size_t)o * CC;
  const float* pr = proxy + (size_t)nn * CC * KOBJ + j;
  float s = 0.f;
  for (int c = 0; c < CC; ++c) s += wr[c] * pr[(size_t)c * KOBJ];
  const float sc  = g[o] * rsqrtf(v[o] + 1e-5f);
  const float val = s * sc + (b[o] - m[o] * sc);
  vb[idx] = fmaxf(val, 0.f);
}

// ---------------- host launcher --------------------------------------------
extern "C" void kernel_launch(void* const* d_in, const int* in_sizes, int n_in,
                              void* d_out, int out_size, void* d_ws, size_t ws_size,
                              hipStream_t stream) {
  const float* x     = (const float*)d_in[0];
  const float* proxy = (const float*)d_in[1];
  const float* wp1 = (const float*)d_in[2];
  const float* wp2 = (const float*)d_in[3];
  const float* wo1 = (const float*)d_in[4];
  const float* wo2 = (const float*)d_in[5];
  const float* wd  = (const float*)d_in[6];
  const float* wu  = (const float*)d_in[7];
  const float* gp1 = (const float*)d_in[8];
  const float* bp1 = (const float*)d_in[9];
  const float* mp1 = (const float*)d_in[10];
  const float* vp1 = (const float*)d_in[11];
  const float* gp2 = (const float*)d_in[12];
  const float* bp2 = (const float*)d_in[13];
  const float* mp2 = (const float*)d_in[14];
  const float* vp2 = (const float*)d_in[15];
  const float* go1 = (const float*)d_in[16];
  const float* bo1 = (const float*)d_in[17];
  const float* mo1 = (const float*)d_in[18];
  const float* vo1 = (const float*)d_in[19];
  const float* go2 = (const float*)d_in[20];
  const float* bo2 = (const float*)d_in[21];
  const float* mo2 = (const float*)d_in[22];
  const float* vo2 = (const float*)d_in[23];
  const float* gd  = (const float*)d_in[24];
  const float* bd  = (const float*)d_in[25];
  const float* md  = (const float*)d_in[26];
  const float* vd  = (const float*)d_in[27];
  const float* gu  = (const float*)d_in[28];
  const float* bu  = (const float*)d_in[29];
  const float* mu  = (const float*)d_in[30];
  const float* vu  = (const float*)d_in[31];

  char* ws = (char*)d_ws;
  const size_t WBYTES = (size_t)CC * CC * 2;            // 512 KB per f16 weight
  const size_t PBYTES = (size_t)NBATCH * KOBJ * CC * 4; // 311 KB per proxy buf
  _Float16* wh1 = (_Float16*)(ws);
  _Float16* wh2 = (_Float16*)(ws + WBYTES);
  _Float16* whu = (_Float16*)(ws + 2 * WBYTES);
  float* b1  = (float*)(ws + 3 * WBYTES);
  float* b2  = (float*)(ws + 3 * WBYTES + 2048);
  float* buf = (float*)(ws + 3 * WBYTES + 4096);
  float* h1  = (float*)(ws + 3 * WBYTES + 6144);
  float* kb  = (float*)(ws + 3 * WBYTES + 6144 + PBYTES);
  float* vb  = (float*)(ws + 3 * WBYTES + 6144 + 2 * PBYTES);

  (void)in_sizes; (void)n_in; (void)out_size; (void)ws_size;

  hipFuncSetAttribute((const void*)oab_main,
                      hipFuncAttributeMaxDynamicSharedMemorySize, SMEM_BYTES);

  const int WG = (CC * CC + 255) / 256;                  // 1024
  fold_weights<<<WG, 256, 0, stream>>>(wp1, gp1, bp1, mp1, vp1, wh1, b1);
  fold_weights<<<WG, 256, 0, stream>>>(wp2, gp2, bp2, mp2, vp2, wh2, b2);
  fold_weights<<<WG, 256, 0, stream>>>(wu,  gu,  bu,  mu,  vu,  whu, buf);

  const int PG = (NBATCH * KOBJ * CC + 255) / 256;       // 304
  proxy_stage1<<<PG, 256, 0, stream>>>(proxy, wo1, go1, bo1, mo1, vo1, h1);
  proxy_stage2<<<PG, 256, 0, stream>>>(h1,    wo2, go2, bo2, mo2, vo2, kb);
  proxy_value <<<PG, 256, 0, stream>>>(proxy, wd,  gd,  bd,  md,  vd,  vb);

  const int nblocks = (NBATCH * HW_) / TP;               // 4096
  oab_main<<<nblocks, 256, SMEM_BYTES, stream>>>(
      x, wh1, b1, wh2, b2, whu, buf, kb, vb, (float*)d_out);
}
